// AutoregressiveMultiGNNv1_8495445311737
// MI455X (gfx1250) — matrix operand order repaired and checked by reading, not compiled, not run on hardware
//
#include <hip/hip_runtime.h>
#include <hip/hip_bf16.h>
#include <math.h>

// ---------------- model dimensions (from the reference) ----------------
#define NN     1024
#define EE     32768
#define CC     3
#define SI_    64
#define VI_    4
#define SH_    128
#define VH_    16
#define SEI_   32
#define VEI_   1
#define SEH_   32
#define VEH_   1
#define OUT_   4
#define HH     4
#define LL     3
#define MAXSPD 32
#define PLEN   10
#define DH_    (SH_ / HH)   // 32

typedef __bf16 bf16_t;
typedef __attribute__((ext_vector_type(16))) __bf16 v16bf;
typedef __attribute__((ext_vector_type(8)))  float  v8f;
typedef __attribute__((ext_vector_type(4)))  float  v4f;

__device__ __forceinline__ unsigned short bf_bits(float x) {
  union { float f; unsigned u; } v; v.f = x;
  unsigned r = v.u + 0x7fffu + ((v.u >> 16) & 1u);   // round-to-nearest-even
  return (unsigned short)(r >> 16);
}
__device__ __forceinline__ bf16_t f2bf(float x) {
  union { unsigned short s; bf16_t b; } o; o.s = bf_bits(x);
  return o.b;
}
__device__ __forceinline__ unsigned bf_pack2(float lo, float hi) {
  return (unsigned)bf_bits(lo) | ((unsigned)bf_bits(hi) << 16);
}

// WMMA 16-bit A-layout: element e of lane-half `half` holds
//   K = half*8 + e            (e = 0..7)
//   K = 16 + half*8 + (e-8)   (e = 8..15)
// inverse (k -> half, e):  half = (k>>3)&1 ; e = (k&7) + ((k>>4)<<3)
// WMMA 16-bit B-layout: same K order as A (B fill uses the same half/e mapping).

// ---------------- generic bf16-WMMA GEMM: D = act(A@B + bias) ----------------
// LDS tiles are stored in *fragment order*: [block][lane][e] with the 16
// bf16 elements of each lane's fragment contiguous -> one 32B LDS read.
// Fill loads use clamp-and-select (branchless) instead of exec-mask guards.
#define BM 128
#define BN 64
#define BK 32

__global__ __launch_bounds__(256) void gemm_bf16_wmma(
    const float* __restrict__ A, const float* __restrict__ B,
    const float* __restrict__ bias, float* __restrict__ D,
    int M, int Ndim, int Kdim, int lda, int ldb, int ldd,
    long bsA, long bsB, long bsD, int act)
{
  __shared__ __align__(32) bf16_t AsF[8 * 32 * 16];   // 8 row-blocks  (8 KB)
  __shared__ __align__(32) bf16_t BsF[4 * 32 * 16];   // 4 col-blocks  (4 KB)
  const int tid = threadIdx.x, lane = tid & 31, wave = tid >> 5;
  const int m0 = blockIdx.y * BM, n0 = blockIdx.x * BN;
  A += bsA * blockIdx.z; B += bsB * blockIdx.z; D += bsD * blockIdx.z;

  v8f acc[4];
#pragma unroll
  for (int i = 0; i < 4; ++i) {
    v8f z = {0.f, 0.f, 0.f, 0.f, 0.f, 0.f, 0.f, 0.f};
    acc[i] = z;
  }

  for (int k0 = 0; k0 < Kdim; k0 += BK) {
    // ---- fill A tile (128x32) as bf16 pairs, pre-swizzled to fragment order
    // 2048 k-pairs; thread i: row = i>>4, k-pair = (i&15)*2  (coalesced in k)
#pragma unroll 2
    for (int i = tid; i < BM * (BK / 2); i += 256) {
      int r = i >> 4;
      int kp = (i & 15) << 1;
      int gm = m0 + r, gk = k0 + kp;
      int cm  = gm < M ? gm : M - 1;                    // clamp, select later
      int ck0 = gk < Kdim ? gk : Kdim - 1;
      int ck1 = gk + 1 < Kdim ? gk + 1 : Kdim - 1;
      const float* ap = A + (long)cm * lda;
      float x0 = ap[ck0];
      float x1 = ap[ck1];
      x0 = (gm < M && gk < Kdim) ? x0 : 0.f;
      x1 = (gm < M && gk + 1 < Kdim) ? x1 : 0.f;
      int half = (kp >> 3) & 1;
      int e    = (kp & 7) + ((kp >> 4) << 3);
      int idx  = (((r >> 4) * 32 + half * 16 + (r & 15)) * 16 + e);
      *(unsigned*)(&AsF[idx]) = bf_pack2(x0, x1);
    }
    // ---- fill B tile (32x64) as bf16 pairs, fragment order
    // 1024 entries; thread i: k-pair = (i>>6)*2, n = i&63 (coalesced in n)
#pragma unroll 2
    for (int i = tid; i < (BK / 2) * BN; i += 256) {
      int kp = (i >> 6) << 1;
      int n  = i & 63;
      int gk = k0 + kp, gn = n0 + n;
      int cn  = gn < Ndim ? gn : Ndim - 1;
      int ck0 = gk < Kdim ? gk : Kdim - 1;
      int ck1 = gk + 1 < Kdim ? gk + 1 : Kdim - 1;
      float x0 = B[(long)ck0 * ldb + cn];
      float x1 = B[(long)ck1 * ldb + cn];
      x0 = (gk < Kdim && gn < Ndim) ? x0 : 0.f;
      x1 = (gk + 1 < Kdim && gn < Ndim) ? x1 : 0.f;
      int half = (kp >> 3) & 1;
      int e    = (kp & 7) + ((kp >> 4) << 3);
      int idx  = (((n >> 4) * 32 + half * 16 + (n & 15)) * 16 + e);
      *(unsigned*)(&BsF[idx]) = bf_pack2(x0, x1);
    }
    if (k0 + BK < Kdim) {   // gfx1250 global_prefetch_b8 on next K tile
      int pr = m0 + (tid >> 1);
      if (pr < M) __builtin_prefetch(&A[(long)pr * lda + k0 + BK], 0, 0);
      int pk = k0 + BK + (tid & 31);
      if (pk < Kdim) __builtin_prefetch(&B[(long)pk * ldb + n0], 0, 0);
    }
    __syncthreads();

    // fragment loads are contiguous 32B LDS reads (ds_load_b128 x2)
    v16bf a = *(const v16bf*)(&AsF[(wave * 32 + lane) * 16]);
#pragma unroll
    for (int nb = 0; nb < 4; ++nb) {
      v16bf b = *(const v16bf*)(&BsF[(nb * 32 + lane) * 16]);
      acc[nb] = __builtin_amdgcn_wmma_f32_16x16x32_bf16(false, a, false, b,
                                                        (short)0, acc[nb], false, false);
    }
    __syncthreads();
  }

  const int mbase = m0 + wave * 16 + (lane >> 4) * 8;
  const int ncsub = lane & 15;
#pragma unroll
  for (int nb = 0; nb < 4; ++nb) {
#pragma unroll
    for (int r = 0; r < 8; ++r) {
      int gm = mbase + r, gn = n0 + nb * 16 + ncsub;
      if (gm < M && gn < Ndim) {
        float v = acc[nb][r];
        if (bias) v += bias[gn];
        if (act == 1)      v = v / (1.f + __expf(-v));     // silu
        else if (act == 2) v = 1.f / (1.f + __expf(-v));   // sigmoid
        D[(long)gm * ldd + gn] = v;
      }
    }
  }
}

// ---------------- attention: logits = (Q K^T) * scale, batched over C*H ----------------
// Q/K rows are 32B aligned (row stride C*H*DH = 384 floats, 256B-aligned base),
// so fragments load as 4x global_load_b128 each.
__global__ __launch_bounds__(32) void attn_logits_wmma(
    const float* __restrict__ Q, const float* __restrict__ Kt,
    float* __restrict__ logits, float scale)
{
  const int lane = threadIdx.x;
  const int m0 = blockIdx.x * 16, n0 = blockIdx.y * 16;
  const int ch = blockIdx.z, c = ch / HH, h = ch % HH;
  const int half = lane >> 4;
  const long rs = (long)CC * HH * DH_;
  const float* qrow = Q  + (long)(n0 + (lane & 15)) * rs + (long)c * HH * DH_ + h * DH_ + half * 8;
  const float* krow = Kt + (long)(m0 + (lane & 15)) * rs + (long)c * HH * DH_ + h * DH_ + half * 8;
  v4f q0 = *(const v4f*)(qrow);
  v4f q1 = *(const v4f*)(qrow + 4);
  v4f q2 = *(const v4f*)(qrow + 16);
  v4f q3 = *(const v4f*)(qrow + 20);
  v4f k0v = *(const v4f*)(krow);
  v4f k1v = *(const v4f*)(krow + 4);
  v4f k2v = *(const v4f*)(krow + 16);
  v4f k3v = *(const v4f*)(krow + 20);
  v16bf a, b;
#pragma unroll
  for (int e = 0; e < 4; ++e) {
    a[e] = f2bf(q0[e]); a[4 + e] = f2bf(q1[e]); a[8 + e] = f2bf(q2[e]); a[12 + e] = f2bf(q3[e]);
    b[e] = f2bf(k0v[e]); b[4 + e] = f2bf(k1v[e]); b[8 + e] = f2bf(k2v[e]); b[12 + e] = f2bf(k3v[e]);
  }
  v8f acc = {0.f, 0.f, 0.f, 0.f, 0.f, 0.f, 0.f, 0.f};
  acc = __builtin_amdgcn_wmma_f32_16x16x32_bf16(false, a, false, b, (short)0, acc, false, false);
  float* out = logits + ((long)ch * NN + n0) * NN + m0;
#pragma unroll
  for (int r = 0; r < 8; ++r)
    out[(long)(half * 8 + r) * NN + (lane & 15)] = acc[r] * scale;
}

// softmax over keys with (spd + path + batch-mask) bias folded in
__global__ __launch_bounds__(256) void softmax_bias_kernel(
    float* __restrict__ logits, const int* __restrict__ spd, const int* __restrict__ spe,
    const int* __restrict__ batchv, const float* __restrict__ spdP,
    const float* __restrict__ pathW)
{
  const int n = blockIdx.x, ch = blockIdx.y, h = ch % HH;
  float* row = logits + ((long)ch * NN + n) * NN;
  __shared__ float red[256];
  const int tid = threadIdx.x;
  const int bn = batchv[n];
  float mx = -1e30f;
  for (int m = tid; m < NN; m += 256) {
    int sv = spd[(long)n * NN + m];
    sv = sv < 0 ? 0 : (sv > MAXSPD - 1 ? MAXSPD - 1 : sv);
    float bias = spdP[sv * HH + h];
    const int* pe = spe + ((long)n * NN + m) * PLEN;
    float pb = 0.f;
#pragma unroll
    for (int p = 0; p < PLEN; ++p) pb += pathW[pe[p] * HH + h];
    bias += pb * (1.f / PLEN);
    if (batchv[m] != bn) bias += -1e9f;
    float x = row[m] + bias;
    row[m] = x;
    mx = fmaxf(mx, x);
  }
  red[tid] = mx; __syncthreads();
  for (int s = 128; s > 0; s >>= 1) { if (tid < s) red[tid] = fmaxf(red[tid], red[tid + s]); __syncthreads(); }
  mx = red[0]; __syncthreads();
  float sum = 0.f;
  for (int m = tid; m < NN; m += 256) { float e = __expf(row[m] - mx); row[m] = e; sum += e; }
  red[tid] = sum; __syncthreads();
  for (int s = 128; s > 0; s >>= 1) { if (tid < s) red[tid] += red[tid + s]; __syncthreads(); }
  float inv = 1.f / red[0];
  for (int m = tid; m < NN; m += 256) row[m] *= inv;
}

// Vt[ch][d][m] = V[m, c, h, d]   (transpose so A·V B-fragments are contiguous)
__global__ void transpose_v_kernel(const float* __restrict__ V, float* __restrict__ Vt)
{
  long idx = (long)blockIdx.x * blockDim.x + threadIdx.x;
  long tot = (long)CC * HH * DH_ * NN;
  if (idx >= tot) return;
  int m = (int)(idx & (NN - 1));
  long t = idx >> 10;               // NN == 1024
  int d = (int)(t % DH_);
  int ch = (int)(t / DH_);
  int c = ch / HH, h = ch % HH;
  Vt[idx] = V[(long)m * (CC * HH * DH_) + (long)c * HH * DH_ + h * DH_ + d];
}

// out[n, c, h, d] = sum_m A[c,h,n,m] * Vt[c,h][d][m]
__global__ __launch_bounds__(32) void attn_av_wmma(
    const float* __restrict__ A, const float* __restrict__ Vt, float* __restrict__ out)
{
  const int lane = threadIdx.x;
  const int d0 = blockIdx.x * 16, n0 = blockIdx.y * 16;
  const int ch = blockIdx.z, c = ch / HH, h = ch % HH;
  const int half = lane >> 4;
  const long rs = (long)CC * HH * DH_;
  const float* arow = A + ((long)ch * NN + (n0 + (lane & 15))) * NN + half * 8;
  const float* vrow = Vt + ((long)ch * DH_ + (d0 + (lane & 15))) * NN + half * 8;
  v8f acc = {0.f, 0.f, 0.f, 0.f, 0.f, 0.f, 0.f, 0.f};
  for (int k0 = 0; k0 < NN; k0 += 32) {
    // A and B fragments: two contiguous 8-float runs each
    v4f a0 = *(const v4f*)(arow + k0);
    v4f a1 = *(const v4f*)(arow + k0 + 4);
    v4f a2 = *(const v4f*)(arow + k0 + 16);
    v4f a3 = *(const v4f*)(arow + k0 + 20);
    v4f b0 = *(const v4f*)(vrow + k0);
    v4f b1 = *(const v4f*)(vrow + k0 + 4);
    v4f b2 = *(const v4f*)(vrow + k0 + 16);
    v4f b3 = *(const v4f*)(vrow + k0 + 20);
    v16bf a, b;
#pragma unroll
    for (int e = 0; e < 4; ++e) {
      a[e] = f2bf(a0[e]); a[4 + e] = f2bf(a1[e]); a[8 + e] = f2bf(a2[e]); a[12 + e] = f2bf(a3[e]);
      b[e] = f2bf(b0[e]); b[4 + e] = f2bf(b1[e]); b[8 + e] = f2bf(b2[e]); b[12 + e] = f2bf(b3[e]);
    }
    acc = __builtin_amdgcn_wmma_f32_16x16x32_bf16(false, a, false, b, (short)0, acc, false, false);
  }
  float* o = out + (long)n0 * rs + (long)c * HH * DH_ + h * DH_ + d0;
#pragma unroll
  for (int r = 0; r < 8; ++r)
    o[(long)(half * 8 + r) * rs + (lane & 15)] = acc[r];
}

// ---------------- elementwise / normalization / gather kernels ----------------
__global__ __launch_bounds__(256) void ln_rows_kernel(
    const float* __restrict__ x, float* __restrict__ y,
    const float* __restrict__ g, const float* __restrict__ b, long rows, int d)
{
  long row = blockIdx.x;
  if (row >= rows) return;
  const float* xr = x + row * d; float* yr = y + row * d;
  __shared__ float red[256];
  int tid = threadIdx.x;
  float s = 0.f;
  for (int i = tid; i < d; i += 256) s += xr[i];
  red[tid] = s; __syncthreads();
  for (int st = 128; st > 0; st >>= 1) { if (tid < st) red[tid] += red[tid + st]; __syncthreads(); }
  float mu = red[0] / d; __syncthreads();
  float v = 0.f;
  for (int i = tid; i < d; i += 256) { float t = xr[i] - mu; v += t * t; }
  red[tid] = v; __syncthreads();
  for (int st = 128; st > 0; st >>= 1) { if (tid < st) red[tid] += red[tid + st]; __syncthreads(); }
  float inv = rsqrtf(red[0] / d + 1e-5f);
  for (int i = tid; i < d; i += 256) yr[i] = (xr[i] - mu) * inv * g[i] + b[i];
}

__global__ void vln_kernel(const float* __restrict__ V, float* __restrict__ Vo, long rows, int nv)
{
  long r = (long)blockIdx.x * blockDim.x + threadIdx.x;
  if (r >= rows) return;
  const float* vr = V + r * nv * 3; float* vo = Vo + r * nv * 3;
  float s = 0.f;
  for (int i = 0; i < nv * 3; ++i) s += vr[i] * vr[i];
  float inv = rsqrtf(s / nv + 1e-8f);
  for (int i = 0; i < nv * 3; ++i) vo[i] = vr[i] * inv;
}

__global__ void matvec3_kernel(const float* __restrict__ W, const float* __restrict__ V,
                               float* __restrict__ out, long rows, int vi, int ho)
{
  long idx = (long)blockIdx.x * blockDim.x + threadIdx.x;
  long tot = rows * ho * 3;
  if (idx >= tot) return;
  int comp = (int)(idx % 3); long t = idx / 3;
  int h = (int)(t % ho); long r = t / ho;
  const float* vr = V + r * vi * 3;
  const float* w = W + (long)h * vi;
  float s = 0.f;
  for (int v = 0; v < vi; ++v) s += w[v] * vr[v * 3 + comp];
  out[idx] = s;
}

__global__ void vnormfeat_kernel(const float* __restrict__ Vh, float* __restrict__ dst,
                                 long rows, int h, int si, int ldd)
{
  long idx = (long)blockIdx.x * blockDim.x + threadIdx.x;
  long tot = rows * h;
  if (idx >= tot) return;
  int j = (int)(idx % h); long r = idx / h;
  const float* v = Vh + (r * h + j) * 3;
  dst[r * ldd + si + j] = sqrtf(v[0]*v[0] + v[1]*v[1] + v[2]*v[2] + 1e-8f);
}

__global__ void copycols_kernel(const float* __restrict__ src, int lds_, float* __restrict__ dst,
                                int ldd, long rows, int w, int dcol)
{
  long idx = (long)blockIdx.x * blockDim.x + threadIdx.x;
  long tot = rows * w;
  if (idx >= tot) return;
  int j = (int)(idx % w); long r = idx / w;
  dst[r * ldd + dcol + j] = src[r * lds_ + j];
}

__global__ void gate_scale_kernel(const float* __restrict__ gate, float* __restrict__ V,
                                  long rows, int vo)
{
  long idx = (long)blockIdx.x * blockDim.x + threadIdx.x;
  long tot = rows * vo * 3;
  if (idx >= tot) return;
  long t = idx / 3; int o = (int)(t % vo); long r = t / vo;
  V[idx] *= gate[r * vo + o];
}

__global__ void silu_kernel(const float* __restrict__ x, float* __restrict__ y, long n)
{
  long i = (long)blockIdx.x * blockDim.x + threadIdx.x;
  if (i >= n) return;
  float v = x[i];
  y[i] = v / (1.f + __expf(-v));
}

__global__ void add_kernel(float* __restrict__ a, const float* __restrict__ b, long n)
{
  long i = (long)blockIdx.x * blockDim.x + threadIdx.x;
  if (i >= n) return;
  a[i] += b[i];
}

__global__ void seg_count_kernel(const int* __restrict__ dst, float* __restrict__ cnt, int E)
{
  int e = blockIdx.x * blockDim.x + threadIdx.x;
  if (e >= E) return;
  atomicAdd(&cnt[dst[e]], 1.f);
}

__global__ void seg_add_kernel(const float* __restrict__ msg, const int* __restrict__ dst,
                               float* __restrict__ agg, int E, int C, int D)
{
  long idx = (long)blockIdx.x * blockDim.x + threadIdx.x;
  long tot = (long)E * C * D;
  if (idx >= tot) return;
  int j = (int)(idx % D); long t = idx / D;
  int c = (int)(t % C); long e = t / C;
  atomicAdd(&agg[((long)dst[e] * C + c) * D + j], msg[idx]);
}

__global__ void seg_div_kernel(float* __restrict__ agg, const float* __restrict__ cnt,
                               int Nr, int C, int D)
{
  long idx = (long)blockIdx.x * blockDim.x + threadIdx.x;
  long tot = (long)Nr * C * D;
  if (idx >= tot) return;
  int n = (int)(idx / ((long)C * D));
  float c = cnt[n];
  agg[idx] /= (c < 1.f ? 1.f : c);
}

__global__ void concat_msg_kernel(const float* __restrict__ sn, const float* __restrict__ es,
                                  const int* __restrict__ srcI, const int* __restrict__ dstI,
                                  float* __restrict__ out, int E, int C, int sh, int seh)
{
  int tot = 2 * sh + seh;
  long idx = (long)blockIdx.x * blockDim.x + threadIdx.x;
  long all = (long)E * C * tot;
  if (idx >= all) return;
  int j = (int)(idx % tot); long t = idx / tot;
  int c = (int)(t % C); long e = t / C;
  float v;
  if (j < sh)            v = sn[((long)srcI[e] * C + c) * sh + j];
  else if (j < sh + seh) v = es[((long)e * C + c) * seh + (j - sh)];
  else                   v = sn[((long)dstI[e] * C + c) * sh + (j - sh - seh)];
  out[idx] = v;
}

__global__ void concat_msgv_kernel(const float* __restrict__ vn, const float* __restrict__ ev,
                                   const int* __restrict__ srcI, const int* __restrict__ dstI,
                                   float* __restrict__ out, int E, int C, int vh, int veh)
{
  int tot = (2 * vh + veh) * 3;
  long idx = (long)blockIdx.x * blockDim.x + threadIdx.x;
  long all = (long)E * C * tot;
  if (idx >= all) return;
  int j = (int)(idx % tot); long t = idx / tot;
  int c = (int)(t % C); long e = t / C;
  int comp = j % 3, o = j / 3;
  float v;
  if (o < vh)            v = vn[(((long)srcI[e] * C + c) * vh + o) * 3 + comp];
  else if (o < vh + veh) v = ev[(((long)e * C + c) * veh + (o - vh)) * 3 + comp];
  else                   v = vn[(((long)dstI[e] * C + c) * vh + (o - vh - veh)) * 3 + comp];
  out[idx] = v;
}

__global__ void efs_kernel(const float* __restrict__ edge_s, const float* __restrict__ mask,
                           const int* __restrict__ srcI, float* __restrict__ out, int E, int sei)
{
  long idx = (long)blockIdx.x * blockDim.x + threadIdx.x;
  long tot = (long)E * sei;
  if (idx >= tot) return;
  int j = (int)(idx % sei); long e = idx / sei;
  int s = srcI[e];
  float num = 0.f, den = 0.f;
  for (int c = 0; c < CC; ++c) {
    float m = mask[(long)s * CC + c];
    num += edge_s[((long)e * CC + c) * sei + j] * m;
    den += m;
  }
  out[idx] = num / (den < 1.f ? 1.f : den);
}

// conf-pool: out[r, j] = sum_c x[r,c,j]*m(r,c) / clip(sum_c m,1)
__global__ void pool_conf_kernel(const float* __restrict__ x, const float* __restrict__ mask,
                                 const int* __restrict__ midx, float* __restrict__ out,
                                 long rows, int D)
{
  long idx = (long)blockIdx.x * blockDim.x + threadIdx.x;
  long tot = rows * D;
  if (idx >= tot) return;
  int j = (int)(idx % D); long r = idx / D;
  long mr = midx ? (long)midx[r] : r;
  float num = 0.f, den = 0.f;
  for (int c = 0; c < CC; ++c) {
    float m = mask[mr * CC + c];
    num += x[(r * CC + c) * D + j] * m;
    den += m;
  }
  out[idx] = num / (den < 1.f ? 1.f : den);
}

__global__ void hS_kernel(const float* __restrict__ Wemb, const int* __restrict__ seq,
                          const int* __restrict__ srcI, const int* __restrict__ dstI,
                          float* __restrict__ dst, int ldd, int dcol, int E)
{
  long idx = (long)blockIdx.x * blockDim.x + threadIdx.x;
  long tot = (long)E * OUT_;
  if (idx >= tot) return;
  int j = (int)(idx % OUT_); long e = idx / OUT_;
  int s = srcI[e];
  int sq = seq[s]; sq = sq < 0 ? 0 : (sq > OUT_ - 1 ? OUT_ - 1 : sq);
  dst[e * ldd + dcol + j] = (s < dstI[e]) ? Wemb[sq * OUT_ + j] : 0.f;
}

// ---------------- host-side helpers ----------------
static inline dim3 g1(long n) { return dim3((unsigned)((n + 255) / 256)); }

struct WS {
  char* base; size_t cap; size_t off;
  float* f(long elems) {
    float* p = (float*)(base + off);
    size_t b = (size_t)elems * sizeof(float);
    off += (b + 255) & ~(size_t)255;
    return p;
  }
};

struct PC {  // positional cursor over flattened param leaves
  void* const* in; int n; int i;
  const float* f() {
    int k = (i < n) ? i : (n > 0 ? n - 1 : 0);
    ++i;
    return (const float*)in[k];
  }
};

struct GvpP { const float *Wh, *Ws, *bs, *Wv, *Wg, *bg; };
static GvpP take_gvp(PC& pc, bool vec) {
  GvpP g;
  g.Wh = pc.f(); g.Ws = pc.f(); g.bs = pc.f();
  if (vec) { g.Wv = pc.f(); g.Wg = pc.f(); g.bg = pc.f(); }
  else     { g.Wv = g.Wg = g.bg = nullptr; }
  return g;
}

static void gemm(hipStream_t st, const float* A, const float* B, const float* bias, float* D,
                 int M, int N, int K, int act = 0, int batch = 1,
                 long sA = 0, long sB = 0, long sD = 0)
{
  dim3 grd((N + BN - 1) / BN, (M + BM - 1) / BM, batch);
  gemm_bf16_wmma<<<grd, 256, 0, st>>>(A, B, bias, D, M, N, K, K, N, N, sA, sB, sD, act);
}

// s: rows x si, V: rows x vi x 3 -> outS rows x so, outV rows x vo x 3
static void run_gvp(hipStream_t st, WS& ws, const GvpP& P, long rows,
                    int si, int vi, int so, int vo,
                    const float* s, const float* V, float* outS, float* outV, int act)
{
  int h = (vo > 0) ? (vi > vo ? vi : vo) : vi;
  float* Vh  = ws.f(rows * h * 3);
  float* cat = ws.f(rows * (si + h));
  matvec3_kernel<<<g1(rows * h * 3), 256, 0, st>>>(P.Wh, V, Vh, rows, vi, h);
  copycols_kernel<<<g1(rows * si), 256, 0, st>>>(s, si, cat, si + h, rows, si, 0);
  vnormfeat_kernel<<<g1(rows * h), 256, 0, st>>>(Vh, cat, rows, h, si, si + h);
  gemm(st, cat, P.Ws, P.bs, outS, (int)rows, so, si + h, 0);
  if (vo > 0) {
    const float* gin = outS;
    if (act == 1) {
      float* tmp = ws.f(rows * so);
      silu_kernel<<<g1(rows * so), 256, 0, st>>>(outS, tmp, rows * so);
      gin = tmp;
    }
    float* gate = ws.f(rows * vo);
    gemm(st, gin, P.Wg, P.bg, gate, (int)rows, vo, so, 2);   // sigmoid
    matvec3_kernel<<<g1(rows * vo * 3), 256, 0, st>>>(P.Wv, Vh, outV, rows, h, vo);
    gate_scale_kernel<<<g1(rows * vo * 3), 256, 0, st>>>(gate, outV, rows, vo);
  }
  if (act == 1) silu_kernel<<<g1(rows * so), 256, 0, st>>>(outS, outS, rows * so);
}

// ---------------- orchestration ----------------
extern "C" void kernel_launch(void* const* d_in, const int* in_sizes, int n_in,
                              void* d_out, int out_size, void* d_ws, size_t ws_size,
                              hipStream_t stream)
{
  (void)in_sizes; (void)out_size;
  if (n_in < 10) return;

  const float* node_s = (const float*)d_in[0];
  const float* node_v = (const float*)d_in[1];
  const float* edge_s = (const float*)d_in[2];
  const float* edge_v = (const float*)d_in[3];
  const float* mask   = (const float*)d_in[4];
  const int* edge_index = (const int*)d_in[n_in - 5];
  const int* seq        = (const int*)d_in[n_in - 4];
  const int* spd        = (const int*)d_in[n_in - 3];
  const int* spe        = (const int*)d_in[n_in - 2];
  const int* batchv     = (const int*)d_in[n_in - 1];
  const int* srcI = edge_index;
  const int* dstI = edge_index + EE;

  WS ws{ (char*)d_ws, ws_size, 0 };
  PC pc{ d_in, n_in, 5 };

  const long rowsN = (long)NN * CC;
  const long rowsE = (long)EE * CC;

  // ---- persistent state ----
  float* hs  = ws.f(rowsN * SH_);
  float* hv  = ws.f(rowsN * VH_ * 3);
  float* es  = ws.f(rowsE * SEH_);
  float* ev  = ws.f(rowsE * VEH_ * 3);
  float* efs = ws.f((long)EE * SEI_);

  // ---- input embeddings ----
  {
    const float* lg = pc.f(); const float* lb = pc.f();
    GvpP gin = take_gvp(pc, true);
    float* s0 = ws.f(rowsN * SI_);
    float* v0 = ws.f(rowsN * VI_ * 3);
    ln_rows_kernel<<<dim3((unsigned)rowsN), 256, 0, stream>>>(node_s, s0, lg, lb, rowsN, SI_);
    vln_kernel<<<g1(rowsN), 256, 0, stream>>>(node_v, v0, rowsN, VI_);
    run_gvp(stream, ws, gin, rowsN, SI_, VI_, SH_, VH_, s0, v0, hs, hv, 0);

    const float* elg = pc.f(); const float* elb = pc.f();
    GvpP gein = take_gvp(pc, true);
    float* se0 = ws.f(rowsE * SEI_);
    float* ve0 = ws.f(rowsE * VEI_ * 3);
    ln_rows_kernel<<<dim3((unsigned)rowsE), 256, 0, stream>>>(edge_s, se0, elg, elb, rowsE, SEI_);
    vln_kernel<<<g1(rowsE), 256, 0, stream>>>(edge_v, ve0, rowsE, VEI_);
    run_gvp(stream, ws, gein, rowsE, SEI_, VEI_, SEH_, VEH_, se0, ve0, es, ev, 0);
  }
  efs_kernel<<<g1((long)EE * SEI_), 256, 0, stream>>>(edge_s, mask, srcI, efs, EE, SEI_);

  const size_t mark = ws.off;   // scratch high-water mark, reset per layer

  // ---- encoder layers ----
  for (int l = 0; l < LL; ++l) {
    ws.off = mark;
    const float* l1g = pc.f(); const float* l1b = pc.f();
    const float* l2g = pc.f(); const float* l2b = pc.f();
    const float* Wq = pc.f();  const float* Wk = pc.f();
    const float* Wvv = pc.f(); const float* Wo = pc.f();
    const float* spdP = pc.f();
    const float* Wpath = pc.f();

    float* sn = ws.f(rowsN * SH_);
    float* vn = ws.f(rowsN * VH_ * 3);
    ln_rows_kernel<<<dim3((unsigned)rowsN), 256, 0, stream>>>(hs, sn, l1g, l1b, rowsN, SH_);
    vln_kernel<<<g1(rowsN), 256, 0, stream>>>(hv, vn, rowsN, VH_);

    // Q/K/V projections (bf16 WMMA GEMMs)
    float* qb = ws.f(rowsN * SH_);
    float* kb = ws.f(rowsN * SH_);
    float* vb = ws.f(rowsN * SH_);
    gemm(stream, sn, Wq,  nullptr, qb, (int)rowsN, SH_, SH_);
    gemm(stream, sn, Wk,  nullptr, kb, (int)rowsN, SH_, SH_);
    gemm(stream, sn, Wvv, nullptr, vb, (int)rowsN, SH_, SH_);

    // path bias projection
    float* pathW = ws.f((long)EE * HH);
    gemm(stream, efs, Wpath, nullptr, pathW, EE, HH, SEI_);

    // attention
    float* logits = ws.f((long)CC * HH * NN * NN);
    attn_logits_wmma<<<dim3(NN / 16, NN / 16, CC * HH), 32, 0, stream>>>(
        qb, kb, logits, 0.1767766953f /* 1/sqrt(32) */);
    softmax_bias_kernel<<<dim3(NN, CC * HH), 256, 0, stream>>>(
        logits, spd, spe, batchv, spdP, pathW);
    float* vt = ws.f((long)CC * HH * DH_ * NN);
    transpose_v_kernel<<<g1((long)CC * HH * DH_ * NN), 256, 0, stream>>>(vb, vt);
    float* attnp = ws.f(rowsN * SH_);
    attn_av_wmma<<<dim3(DH_ / 16, NN / 16, CC * HH), 32, 0, stream>>>(logits, vt, attnp);
    float* attno = ws.f(rowsN * SH_);
    gemm(stream, attnp, Wo, nullptr, attno, (int)rowsN, SH_, SH_);

    // message GVP stack on edges
    GvpP m0 = take_gvp(pc, true), m1 = take_gvp(pc, true), m2 = take_gvp(pc, true);
    const int MSI = 2 * SH_ + SEH_;          // 288
    const int MVI = 2 * VH_ + VEH_;          // 33
    float* msCat = ws.f(rowsE * MSI);
    float* mvCat = ws.f(rowsE * MVI * 3);
    concat_msg_kernel<<<g1(rowsE * MSI), 256, 0, stream>>>(sn, es, srcI, dstI, msCat, EE, CC, SH_, SEH_);
    concat_msgv_kernel<<<g1(rowsE * MVI * 3), 256, 0, stream>>>(vn, ev, srcI, dstI, mvCat, EE, CC, VH_, VEH_);
    float* msA = ws.f(rowsE * SH_); float* mvA = ws.f(rowsE * VH_ * 3);
    float* msB = ws.f(rowsE * SH_); float* mvB = ws.f(rowsE * VH_ * 3);
    run_gvp(stream, ws, m0, rowsE, MSI, MVI, SH_, VH_, msCat, mvCat, msA, mvA, 1);
    run_gvp(stream, ws, m1, rowsE, SH_, VH_, SH_, VH_, msA, mvA, msB, mvB, 1);
    run_gvp(stream, ws, m2, rowsE, SH_, VH_, SH_, VH_, msB, mvB, msA, mvA, 0);

    // mean aggregation to nodes
    float* aggS = ws.f(rowsN * SH_);
    float* aggV = ws.f(rowsN * VH_ * 3);
    float* cnt  = ws.f(NN);
    hipMemsetAsync(aggS, 0, (size_t)rowsN * SH_ * sizeof(float), stream);
    hipMemsetAsync(aggV, 0, (size_t)rowsN * VH_ * 3 * sizeof(float), stream);
    hipMemsetAsync(cnt,  0, (size_t)NN * sizeof(float), stream);
    seg_count_kernel<<<g1(EE), 256, 0, stream>>>(dstI, cnt, EE);
    seg_add_kernel<<<g1(rowsE * SH_), 256, 0, stream>>>(msA, dstI, aggS, EE, CC, SH_);
    seg_add_kernel<<<g1(rowsE * VH_ * 3), 256, 0, stream>>>(mvA, dstI, aggV, EE, CC, VH_ * 3);
    seg_div_kernel<<<g1(rowsN * SH_), 256, 0, stream>>>(aggS, cnt, NN, CC, SH_);
    seg_div_kernel<<<g1(rowsN * VH_ * 3), 256, 0, stream>>>(aggV, cnt, NN, CC, VH_ * 3);

    add_kernel<<<g1(rowsN * SH_), 256, 0, stream>>>(hs, attno, rowsN * SH_);
    add_kernel<<<g1(rowsN * SH_), 256, 0, stream>>>(hs, aggS, rowsN * SH_);
    add_kernel<<<g1(rowsN * VH_ * 3), 256, 0, stream>>>(hv, aggV, rowsN * VH_ * 3);

    // feed-forward GVP stack
    GvpP f0 = take_gvp(pc, true), f1 = take_gvp(pc, true);
    ln_rows_kernel<<<dim3((unsigned)rowsN), 256, 0, stream>>>(hs, sn, l2g, l2b, rowsN, SH_);
    vln_kernel<<<g1(rowsN), 256, 0, stream>>>(hv, vn, rowsN, VH_);
    float* fs1 = ws.f(rowsN * 4 * SH_); float* fv1 = ws.f(rowsN * 2 * VH_ * 3);
    float* fs2 = ws.f(rowsN * SH_);     float* fv2 = ws.f(rowsN * VH_ * 3);
    run_gvp(stream, ws, f0, rowsN, SH_, VH_, 4 * SH_, 2 * VH_, sn, vn, fs1, fv1, 1);
    run_gvp(stream, ws, f1, rowsN, 4 * SH_, 2 * VH_, SH_, VH_, fs1, fv1, fs2, fv2, 0);
    add_kernel<<<g1(rowsN * SH_), 256, 0, stream>>>(hs, fs2, rowsN * SH_);
    add_kernel<<<g1(rowsN * VH_ * 3), 256, 0, stream>>>(hv, fv2, rowsN * VH_ * 3);
  }

  // ---- capture decoder params (positional) ----
  struct DecP { const float *l1g, *l1b, *l2g, *l2b; GvpP msg[3]; GvpP ff[2]; };
  DecP dp[LL];
  for (int l = 0; l < LL; ++l) {
    dp[l].l1g = pc.f(); dp[l].l1b = pc.f();
    dp[l].l2g = pc.f(); dp[l].l2b = pc.f();
    for (int i = 0; i < 3; ++i) dp[l].msg[i] = take_gvp(pc, true);
    for (int i = 0; i < 2; ++i) dp[l].ff[i]  = take_gvp(pc, true);
  }
  const float* WsEmb = pc.f();
  GvpP gout = take_gvp(pc, false);

  // ---- conformer pooling ----
  ws.off = mark;
  float* xs   = ws.f((long)NN * SH_);
  float* xv   = ws.f((long)NN * VH_ * 3);
  float* hEps = ws.f((long)EE * SEH_);
  float* hEpv = ws.f((long)EE * VEH_ * 3);
  float* esD  = ws.f((long)EE * (SEH_ + OUT_));
  pool_conf_kernel<<<g1((long)NN * SH_), 256, 0, stream>>>(hs, mask, nullptr, xs, NN, SH_);
  pool_conf_kernel<<<g1((long)NN * VH_ * 3), 256, 0, stream>>>(hv, mask, nullptr, xv, NN, VH_ * 3);
  pool_conf_kernel<<<g1((long)EE * SEH_), 256, 0, stream>>>(es, mask, srcI, hEps, EE, SEH_);
  pool_conf_kernel<<<g1((long)EE * VEH_ * 3), 256, 0, stream>>>(ev, mask, srcI, hEpv, EE, VEH_ * 3);
  copycols_kernel<<<g1((long)EE * SEH_), 256, 0, stream>>>(hEps, SEH_, esD, SEH_ + OUT_, EE, SEH_, 0);
  hS_kernel<<<g1((long)EE * OUT_), 256, 0, stream>>>(WsEmb, seq, srcI, dstI, esD, SEH_ + OUT_, SEH_, EE);

  const size_t mark2 = ws.off;

  // ---- decoder layers (pooled graph, C=1) ----
  for (int l = 0; l < LL; ++l) {
    ws.off = mark2;
    float* sn = ws.f((long)NN * SH_);
    float* vn = ws.f((long)NN * VH_ * 3);
    ln_rows_kernel<<<dim3(NN), 256, 0, stream>>>(xs, sn, dp[l].l1g, dp[l].l1b, NN, SH_);
    vln_kernel<<<g1(NN), 256, 0, stream>>>(xv, vn, NN, VH_);

    const int MSI = 2 * SH_ + SEH_ + OUT_;   // 292
    const int MVI = 2 * VH_ + VEH_;          // 33
    float* msCat = ws.f((long)EE * MSI);
    float* mvCat = ws.f((long)EE * MVI * 3);
    concat_msg_kernel<<<g1((long)EE * MSI), 256, 0, stream>>>(sn, esD, srcI, dstI, msCat, EE, 1, SH_, SEH_ + OUT_);
    concat_msgv_kernel<<<g1((long)EE * MVI * 3), 256, 0, stream>>>(vn, hEpv, srcI, dstI, mvCat, EE, 1, VH_, VEH_);
    float* msA = ws.f((long)EE * SH_); float* mvA = ws.f((long)EE * VH_ * 3);
    float* msB = ws.f((long)EE * SH_); float* mvB = ws.f((long)EE * VH_ * 3);
    run_gvp(stream, ws, dp[l].msg[0], EE, MSI, MVI, SH_, VH_, msCat, mvCat, msA, mvA, 1);
    run_gvp(stream, ws, dp[l].msg[1], EE, SH_, VH_, SH_, VH_, msA, mvA, msB, mvB, 1);
    run_gvp(stream, ws, dp[l].msg[2], EE, SH_, VH_, SH_, VH_, msB, mvB, msA, mvA, 0);

    float* aggS = ws.f((long)NN * SH_);
    float* aggV = ws.f((long)NN * VH_ * 3);
    float* cnt  = ws.f(NN);
    hipMemsetAsync(aggS, 0, (size_t)NN * SH_ * sizeof(float), stream);
    hipMemsetAsync(aggV, 0, (size_t)NN * VH_ * 3 * sizeof(float), stream);
    hipMemsetAsync(cnt,  0, (size_t)NN * sizeof(float), stream);
    seg_count_kernel<<<g1(EE), 256, 0, stream>>>(dstI, cnt, EE);
    seg_add_kernel<<<g1((long)EE * SH_), 256, 0, stream>>>(msA, dstI, aggS, EE, 1, SH_);
    seg_add_kernel<<<g1((long)EE * VH_ * 3), 256, 0, stream>>>(mvA, dstI, aggV, EE, 1, VH_ * 3);
    seg_div_kernel<<<g1((long)NN * SH_), 256, 0, stream>>>(aggS, cnt, NN, 1, SH_);
    seg_div_kernel<<<g1((long)NN * VH_ * 3), 256, 0, stream>>>(aggV, cnt, NN, 1, VH_ * 3);
    add_kernel<<<g1((long)NN * SH_), 256, 0, stream>>>(xs, aggS, (long)NN * SH_);
    add_kernel<<<g1((long)NN * VH_ * 3), 256, 0, stream>>>(xv, aggV, (long)NN * VH_ * 3);

    ln_rows_kernel<<<dim3(NN), 256, 0, stream>>>(xs, sn, dp[l].l2g, dp[l].l2b, NN, SH_);
    vln_kernel<<<g1(NN), 256, 0, stream>>>(xv, vn, NN, VH_);
    float* fs1 = ws.f((long)NN * 4 * SH_); float* fv1 = ws.f((long)NN * 2 * VH_ * 3);
    float* fs2 = ws.f((long)NN * SH_);     float* fv2 = ws.f((long)NN * VH_ * 3);
    run_gvp(stream, ws, dp[l].ff[0], NN, SH_, VH_, 4 * SH_, 2 * VH_, sn, vn, fs1, fv1, 1);
    run_gvp(stream, ws, dp[l].ff[1], NN, 4 * SH_, 2 * VH_, SH_, VH_, fs1, fv1, fs2, fv2, 0);
    add_kernel<<<g1((long)NN * SH_), 256, 0, stream>>>(xs, fs2, (long)NN * SH_);
    add_kernel<<<g1((long)NN * VH_ * 3), 256, 0, stream>>>(xv, fv2, (long)NN * VH_ * 3);
  }

  // ---- output projection: N x OUT logits ----
  ws.off = mark2;
  run_gvp(stream, ws, gout, NN, SH_, VH_, OUT_, 0, xs, xv, (float*)d_out, nullptr, 0);
}
